// Attention_51238959841962
// MI455X (gfx1250) — matrix operand order repaired and checked by reading, not compiled, not run on hardware
//
#include <hip/hip_runtime.h>
#include <stdint.h>

typedef __attribute__((ext_vector_type(16))) _Float16 v16h;
typedef __attribute__((ext_vector_type(8)))  float    v8f;

union F16Frag { uint32_t u[8]; v16h v; };

// ---------------------------------------------------------------------------
// WMMA helpers (CDNA5 gfx1250, wave32)
// ---------------------------------------------------------------------------
__device__ __forceinline__ v8f wmma_f16(v16h a, v16h b, v8f c) {
    return __builtin_amdgcn_wmma_f32_16x16x32_f16(false, a, false, b,
                                                  (short)0, c, false, false);
}

// A fragment: 16x32 f16, stored row-major [16][ldh] at `base`.
// lane l: row = l&15 ; half h = l>>4
//   VGPR j in 0..3 : K = h*8 + 2j    ; VGPR j in 4..7 : K = 16 + h*8 + 2(j-4)
__device__ __forceinline__ v16h load_a_frag(const _Float16* base, int ldh, int lane) {
    const int row = lane & 15, h = lane >> 4;
    const _Float16* p = base + row * ldh;
    F16Frag f;
#pragma unroll
    for (int j = 0; j < 4; ++j)
        f.u[j] = *(const uint32_t*)(p + h * 8 + 2 * j);
#pragma unroll
    for (int j = 0; j < 4; ++j)
        f.u[4 + j] = *(const uint32_t*)(p + 16 + h * 8 + 2 * j);
    return f.v;
}

// B fragment: 32x16 f16 where we hold B^T row-major [16 cols][ldh] at `base`.
// lane l: col = l&15 ; half h = l>>4
//   VGPR j in 0..3 : K = h*8 + 2j    ; VGPR j in 4..7 : K = 16 + h*8 + 2(j-4)
__device__ __forceinline__ v16h load_b_frag(const _Float16* base, int ldh, int lane) {
    const int col = lane & 15, h = lane >> 4;
    const _Float16* p = base + col * ldh;
    F16Frag f;
#pragma unroll
    for (int j = 0; j < 4; ++j)
        f.u[j] = *(const uint32_t*)(p + h * 8 + 2 * j);
#pragma unroll
    for (int j = 0; j < 4; ++j)
        f.u[4 + j] = *(const uint32_t*)(p + 16 + h * 8 + 2 * j);
    return f.v;
}

// ---------------------------------------------------------------------------
// CDNA5 async global->LDS staging (ASYNCcnt-tracked, no data VGPRs)
// ---------------------------------------------------------------------------
typedef __attribute__((address_space(3))) const _Float16 lds_cf16;

__device__ __forceinline__ uint32_t lds_off(const _Float16* p) {
    // ptrtoint of an addrspace(3) pointer is the wave-relative LDS byte offset,
    // which is what GLOBAL_LOAD_ASYNC_TO_LDS expects in its VDST VGPR.
    return (uint32_t)(uintptr_t)(lds_cf16*)p;
}

__device__ __forceinline__ void async_load_b128(uint32_t lds_addr, const void* gptr) {
    asm volatile("global_load_async_to_lds_b128 %0, %1, off"
                 :: "v"(lds_addr), "v"((uint64_t)(uintptr_t)gptr)
                 : "memory");
}

__device__ __forceinline__ void wait_async0() {
    asm volatile("s_wait_asynccnt 0x0" ::: "memory");
}

// ---------------------------------------------------------------------------
// Kernel 1: GroupNorm.  x[b,512,1024] -> xn f32 (residual) + xnT f16 [b][n][c]
// ---------------------------------------------------------------------------
__global__ void gn_kernel(const float* __restrict__ x,
                          const float* __restrict__ gamma,
                          const float* __restrict__ beta,
                          float* __restrict__ xn,
                          _Float16* __restrict__ xnT) {
    __shared__ float sbuf[8];
    __shared__ float stats[2];
    const int tid = threadIdx.x, w = tid >> 5, lane = tid & 31;
    const int b = blockIdx.x >> 3, g = blockIdx.x & 7;
    const float* xp = x + ((size_t)b * 512 + g * 64) * 1024;

    float s = 0.f, ss = 0.f;
    for (int i = tid; i < 65536; i += 256) {
        float v = xp[i];
        s += v; ss += v * v;
    }
#pragma unroll
    for (int off = 16; off > 0; off >>= 1) { s += __shfl_xor(s, off, 32); ss += __shfl_xor(ss, off, 32); }
    if (lane == 0) sbuf[w] = s;
    __syncthreads();
    if (tid == 0) { float t = 0.f; for (int i = 0; i < 8; ++i) t += sbuf[i]; stats[0] = t; }
    __syncthreads();
    if (lane == 0) sbuf[w] = ss;
    __syncthreads();
    if (tid == 0) { float t = 0.f; for (int i = 0; i < 8; ++i) t += sbuf[i]; stats[1] = t; }
    __syncthreads();

    const float mean = stats[0] * (1.f / 65536.f);
    const float var  = stats[1] * (1.f / 65536.f) - mean * mean;
    const float rstd = rsqrtf(var + 1e-5f);

    for (int i = tid; i < 65536; i += 256) {
        int ch = g * 64 + (i >> 10);
        int pix = i & 1023;
        float v = (xp[i] - mean) * rstd * gamma[ch] + beta[ch];
        xn[((size_t)b * 512 + ch) * 1024 + pix] = v;
        xnT[((size_t)b * 1024 + pix) * 512 + ch] = (_Float16)v;
    }
}

// ---------------------------------------------------------------------------
// Kernel 2: weight f32 -> f16
// ---------------------------------------------------------------------------
__global__ void wconv_kernel(const float* __restrict__ wqkv,
                             const float* __restrict__ wout,
                             _Float16* __restrict__ wqkvh,
                             _Float16* __restrict__ wouth) {
    int i = blockIdx.x * 256 + threadIdx.x;
    if (i < 1536 * 512) wqkvh[i] = (_Float16)wqkv[i];
    if (i < 512 * 512)  wouth[i] = (_Float16)wout[i];
}

// ---------------------------------------------------------------------------
// Shared 128x128 WMMA GEMM mainloop: D = A[M][512] * (Bt[N][512])^T
// 8 waves as 4(M) x 2(N), each wave owns a 32x64 register tile (2x4 frags).
// Double-buffered LDS staging via GLOBAL_LOAD_ASYNC_TO_LDS_B128: tile k+1 is
// fetched asynchronously while tile k feeds the WMMA pipes.
// ---------------------------------------------------------------------------
__device__ __forceinline__ void gemm128_mainloop(const _Float16* __restrict__ A,
                                                 const _Float16* __restrict__ Bt,
                                                 int mBase, int nBase,
                                                 v8f acc[2][4]) {
    __shared__ _Float16 As[2][128 * 48];  // 16-half pad: 16B-aligned, low-conflict
    __shared__ _Float16 Bs[2][128 * 48];
    const int tid = threadIdx.x, w = tid >> 5, lane = tid & 31;
    const int wm = w >> 1, wn = w & 1;
    constexpr int K = 512;

    // each thread owns two 16B chunks of the 128x32 tile
    const int c0 = tid * 2;
    const int row0 = c0 >> 2,      col0 = (c0 & 3) * 8;
    const int row1 = (c0 + 1) >> 2, col1 = ((c0 + 1) & 3) * 8;
    const _Float16* gA0 = A + (size_t)(mBase + row0) * K + col0;
    const _Float16* gA1 = A + (size_t)(mBase + row1) * K + col1;
    const _Float16* gB0 = Bt + (size_t)(nBase + row0) * K + col0;
    const _Float16* gB1 = Bt + (size_t)(nBase + row1) * K + col1;

    // prologue: stage k0 = 0 into buffer 0
    async_load_b128(lds_off(&As[0][row0 * 48 + col0]), gA0);
    async_load_b128(lds_off(&As[0][row1 * 48 + col1]), gA1);
    async_load_b128(lds_off(&Bs[0][row0 * 48 + col0]), gB0);
    async_load_b128(lds_off(&Bs[0][row1 * 48 + col1]), gB1);
    wait_async0();
    __syncthreads();

    for (int k0 = 0; k0 < K; k0 += 32) {
        const int cur = (k0 >> 5) & 1, nxt = cur ^ 1;
        if (k0 + 32 < K) {  // async-prefetch next K tile into the other buffer
            const int kn = k0 + 32;
            async_load_b128(lds_off(&As[nxt][row0 * 48 + col0]), gA0 + kn);
            async_load_b128(lds_off(&As[nxt][row1 * 48 + col1]), gA1 + kn);
            async_load_b128(lds_off(&Bs[nxt][row0 * 48 + col0]), gB0 + kn);
            async_load_b128(lds_off(&Bs[nxt][row1 * 48 + col1]), gB1 + kn);
        }

        v16h a0 = load_a_frag(&As[cur][(wm * 32) * 48], 48, lane);
        v16h a1 = load_a_frag(&As[cur][(wm * 32 + 16) * 48], 48, lane);
#pragma unroll
        for (int jn = 0; jn < 4; ++jn) {
            v16h bf = load_b_frag(&Bs[cur][(wn * 64 + jn * 16) * 48], 48, lane);
            acc[0][jn] = wmma_f16(a0, bf, acc[0][jn]);
            acc[1][jn] = wmma_f16(a1, bf, acc[1][jn]);
        }

        wait_async0();     // own async writes into nxt buffer have landed
        __syncthreads();   // everyone done reading cur / writing nxt
    }
}

// ---------------------------------------------------------------------------
// Kernel 3: QKV GEMM.  qkv[o,n] = Wqkv[o,c] * xn[c,n]  (per batch)
// Scatter epilogue -> q[bh][n][d] (scaled), k[bh][n][d], v[bh][d][n]  (f16)
// ---------------------------------------------------------------------------
__global__ void qkv_gemm_kernel(const _Float16* __restrict__ Wh,   // [1536][512]
                                const _Float16* __restrict__ xnT,  // [b][1024][512]
                                _Float16* __restrict__ qb,
                                _Float16* __restrict__ kb,
                                _Float16* __restrict__ vb) {
    const int tid = threadIdx.x, w = tid >> 5, lane = tid & 31;
    const int wm = w >> 1, wn = w & 1, hh = lane >> 4, lr = lane & 15;
    const int mBase = blockIdx.x * 128, nBase = blockIdx.y * 128, bz = blockIdx.z;

    v8f acc[2][4] = {};
    gemm128_mainloop(Wh, xnT + (size_t)bz * 1024 * 512, mBase, nBase, acc);

#pragma unroll
    for (int i = 0; i < 2; ++i)
#pragma unroll
        for (int jn = 0; jn < 4; ++jn)
#pragma unroll
            for (int r = 0; r < 8; ++r) {
                int Mg = mBase + wm * 32 + i * 16 + hh * 8 + r;
                int Ng = nBase + wn * 64 + jn * 16 + lr;
                float val = acc[i][jn][r];
                int sec = Mg >> 9;           // 0:q 1:k 2:v
                int oc = Mg & 511;
                int head = oc >> 6, dd = oc & 63;
                size_t bh = (size_t)bz * 8 + head;
                if (sec == 0)
                    qb[(bh * 1024 + Ng) * 64 + dd] = (_Float16)(val * 0.125f);  // * d^-0.5
                else if (sec == 1)
                    kb[(bh * 1024 + Ng) * 64 + dd] = (_Float16)val;
                else
                    vb[(bh * 64 + dd) * 1024 + Ng] = (_Float16)val;
            }
}

// ---------------------------------------------------------------------------
// Kernel 4: attention.  One block = 16 query rows of one (b,head).
// Scores for all 1024 keys live in registers (8 v8f frags / wave).
// ---------------------------------------------------------------------------
__global__ void attn_kernel(const _Float16* __restrict__ qb,   // [64][1024][64]
                            const _Float16* __restrict__ kb,   // [64][1024][64]
                            const _Float16* __restrict__ vb,   // [64][64][1024]
                            _Float16* __restrict__ attT) {     // [8][1024][512]
    __shared__ _Float16 qs[16 * 72];          // padded q tile (A operand)
    __shared__ _Float16 attn_s[16 * 1032];    // padded probabilities (A operand)
    __shared__ float outacc[16 * 64];
    __shared__ float redmax[8][16];
    __shared__ float redsum[8][16];

    const int tid = threadIdx.x, w = tid >> 5, lane = tid & 31;
    const int hh = lane >> 4, lr = lane & 15;
    const int i0 = blockIdx.x * 16;
    const int bh = blockIdx.y, b = bh >> 3, head = bh & 7;

    for (int e = tid; e < 1024; e += 256) {
        int rr = e >> 6, dd = e & 63;
        qs[rr * 72 + dd] = qb[((size_t)bh * 1024 + i0 + rr) * 64 + dd];
        outacc[e] = 0.f;
    }
    __syncthreads();

    // --- phase 1: sim = q k^T for this wave's 128 key columns ---
    const int jbase = w * 128;
    v8f sc[8];
#pragma unroll
    for (int nf = 0; nf < 8; ++nf) {
        v8f a = {};
#pragma unroll
        for (int kk = 0; kk < 64; kk += 32) {
            v16h af = load_a_frag(qs + kk, 72, lane);
            v16h bf = load_b_frag(kb + ((size_t)bh * 1024 + jbase + nf * 16) * 64 + kk, 64, lane);
            a = wmma_f16(af, bf, a);
        }
        sc[nf] = a;
    }

    // --- row max (rows hh*8+r): width-16 shuffle then cross-wave LDS ---
    float pm[8];
#pragma unroll
    for (int r = 0; r < 8; ++r) {
        float m = sc[0][r];
#pragma unroll
        for (int nf = 1; nf < 8; ++nf) m = fmaxf(m, sc[nf][r]);
#pragma unroll
        for (int off = 1; off < 16; off <<= 1) m = fmaxf(m, __shfl_xor(m, off, 32));
        pm[r] = m;
    }
    if (lr == 0)
        for (int r = 0; r < 8; ++r) redmax[w][hh * 8 + r] = pm[r];
    __syncthreads();
    float gm[8];
#pragma unroll
    for (int r = 0; r < 8; ++r) {
        float m = redmax[0][hh * 8 + r];
        for (int wv = 1; wv < 8; ++wv) m = fmaxf(m, redmax[wv][hh * 8 + r]);
        gm[r] = m;
    }

    // --- exp + row sum ---
    float ps[8];
#pragma unroll
    for (int r = 0; r < 8; ++r) ps[r] = 0.f;
#pragma unroll
    for (int nf = 0; nf < 8; ++nf)
#pragma unroll
        for (int r = 0; r < 8; ++r) {
            float e = __expf(sc[nf][r] - gm[r]);
            sc[nf][r] = e;
            ps[r] += e;
        }
#pragma unroll
    for (int r = 0; r < 8; ++r) {
        float s = ps[r];
#pragma unroll
        for (int off = 1; off < 16; off <<= 1) s += __shfl_xor(s, off, 32);
        ps[r] = s;
    }
    if (lr == 0)
        for (int r = 0; r < 8; ++r) redsum[w][hh * 8 + r] = ps[r];
    __syncthreads();
    float inv[8];
#pragma unroll
    for (int r = 0; r < 8; ++r) {
        float s = redsum[0][hh * 8 + r];
        for (int wv = 1; wv < 8; ++wv) s += redsum[wv][hh * 8 + r];
        inv[r] = 1.f / s;
    }

    // normalized probabilities -> LDS (A-operand layout, padded stride 1032)
#pragma unroll
    for (int nf = 0; nf < 8; ++nf)
#pragma unroll
        for (int r = 0; r < 8; ++r)
            attn_s[(hh * 8 + r) * 1032 + jbase + nf * 16 + lr] =
                (_Float16)(sc[nf][r] * inv[r]);
    __syncthreads();

    // --- phase 2: out = attn * v^T ; N split over 4 waves, K halved over 2 ---
    const int nt = w & 3, kh = w >> 2;
    v8f oa = {};
    for (int ks = 0; ks < 16; ++ks) {
        int kbk = kh * 512 + ks * 32;
        v16h af = load_a_frag(attn_s + kbk, 1032, lane);
        v16h bf = load_b_frag(vb + ((size_t)bh * 64 + nt * 16) * 1024 + kbk, 1024, lane);
        oa = wmma_f16(af, bf, oa);
    }
#pragma unroll
    for (int r = 0; r < 8; ++r)
        atomicAdd(&outacc[(hh * 8 + r) * 64 + nt * 16 + lr], oa[r]);
    __syncthreads();

    // write [b][n][c=head*64+dd] f16 (B^T layout for the out-proj GEMM)
    for (int e = tid; e < 1024; e += 256) {
        int rr = e >> 6, dd = e & 63;
        attT[((size_t)b * 1024 + i0 + rr) * 512 + head * 64 + dd] = (_Float16)outacc[e];
    }
}

// ---------------------------------------------------------------------------
// Kernel 5: out-proj GEMM + bias + residual -> d_out (fp32)
// ---------------------------------------------------------------------------
__global__ void outproj_kernel(const _Float16* __restrict__ Wh,    // [512][512]
                               const _Float16* __restrict__ attT,  // [b][1024][512]
                               const float* __restrict__ b_out,
                               const float* __restrict__ xn,       // [b][512][1024]
                               float* __restrict__ out) {
    const int tid = threadIdx.x, w = tid >> 5, lane = tid & 31;
    const int wm = w >> 1, wn = w & 1, hh = lane >> 4, lr = lane & 15;
    const int mBase = blockIdx.x * 128, nBase = blockIdx.y * 128, bz = blockIdx.z;

    v8f acc[2][4] = {};
    gemm128_mainloop(Wh, attT + (size_t)bz * 1024 * 512, mBase, nBase, acc);

#pragma unroll
    for (int i = 0; i < 2; ++i)
#pragma unroll
        for (int jn = 0; jn < 4; ++jn)
#pragma unroll
            for (int r = 0; r < 8; ++r) {
                int Mg = mBase + wm * 32 + i * 16 + hh * 8 + r;
                int Ng = nBase + wn * 64 + jn * 16 + lr;
                size_t idx = ((size_t)bz * 512 + Mg) * 1024 + Ng;
                out[idx] = acc[i][jn][r] + b_out[Mg] + xn[idx];
            }
}

// ---------------------------------------------------------------------------
// Host launcher
// ---------------------------------------------------------------------------
extern "C" void kernel_launch(void* const* d_in, const int* in_sizes, int n_in,
                              void* d_out, int out_size, void* d_ws, size_t ws_size,
                              hipStream_t stream) {
    const float* x     = (const float*)d_in[0];
    const float* gamma = (const float*)d_in[1];
    const float* beta  = (const float*)d_in[2];
    const float* w_qkv = (const float*)d_in[3];
    const float* w_out = (const float*)d_in[4];
    const float* b_out = (const float*)d_in[5];
    float* out = (float*)d_out;

    char* p = (char*)d_ws;
    auto carve = [&](size_t bytes) {
        void* r = (void*)p;
        p += (bytes + 255) & ~(size_t)255;
        return r;
    };
    float*     xn    = (float*)    carve((size_t)8 * 512 * 1024 * 4);
    _Float16*  xnT   = (_Float16*) carve((size_t)8 * 1024 * 512 * 2);
    _Float16*  wqkvh = (_Float16*) carve((size_t)1536 * 512 * 2);
    _Float16*  wouth = (_Float16*) carve((size_t)512 * 512 * 2);
    _Float16*  qb    = (_Float16*) carve((size_t)64 * 1024 * 64 * 2);
    _Float16*  kbuf  = (_Float16*) carve((size_t)64 * 1024 * 64 * 2);
    _Float16*  vbuf  = (_Float16*) carve((size_t)64 * 64 * 1024 * 2);
    _Float16*  attT  = (_Float16*) carve((size_t)8 * 1024 * 512 * 2);

    gn_kernel<<<64, 256, 0, stream>>>(x, gamma, beta, xn, xnT);
    wconv_kernel<<<3072, 256, 0, stream>>>(w_qkv, w_out, wqkvh, wouth);
    qkv_gemm_kernel<<<dim3(12, 8, 8), 256, 0, stream>>>(wqkvh, xnT, qb, kbuf, vbuf);
    attn_kernel<<<dim3(64, 64), 256, 0, stream>>>(qb, kbuf, vbuf, attT);
    outproj_kernel<<<dim3(4, 8, 8), 256, 0, stream>>>(wouth, attT, b_out, xn, out);
}
